// ConstaintIBSDE_6786048327949
// MI455X (gfx1250) — compile-verified, matrix operand order checked by reading, not run on hardware
//
#include <hip/hip_runtime.h>
#include <math.h>
#include <stdint.h>

// ---------------- types ----------------
typedef __bf16 bf16_t;
typedef __attribute__((ext_vector_type(16))) __bf16 v16bf;
typedef __attribute__((ext_vector_type(8)))  float  v8f;

#define B_TOT 8192
#define DXP   16
#define DYP   16
#define NSTEP 50
#define HDIM  256
#define DDIM  64
#define BT    32           // batch rows per block
#define DT_F  (1.0f/50.0f)
#define PI_2F 1.57079632679489662f

__device__ __forceinline__ uint16_t f2bf_u(float f) {
  uint32_t u = __builtin_bit_cast(uint32_t, f);
  uint32_t r = u + 0x7FFFu + ((u >> 16) & 1u);   // round-to-nearest-even
  return (uint16_t)(r >> 16);
}
__device__ __forceinline__ bf16_t f2bf(float f) {
  uint16_t h = f2bf_u(f);
  return __builtin_bit_cast(bf16_t, h);
}
__device__ __forceinline__ float bf2f(bf16_t h) {
  uint32_t u = ((uint32_t)__builtin_bit_cast(uint16_t, h)) << 16;
  return __builtin_bit_cast(float, u);
}

// Branch-free tanh: 1 - 2/(exp(2x)+1). v_exp_f32 + v_rcp_f32, no exec
// divergence; saturates correctly for large |x|. Plenty accurate for bf16.
__device__ __forceinline__ float fast_tanh(float x) {
  float ex = __expf(2.f * x);
  return 1.f - 2.f * __builtin_amdgcn_rcpf(1.f + ex);
}

// ---------------- weight pack kernel ----------------
// Packs row-major f32 W[K x N] (nsteps batched) into bf16 WMMA B-fragments.
// Fragment (kt,nt) = 32x16 tile; per lane 16 bf16 (32B) in exact B-operand
// VGPR order: lane n (0..15) = column nt*16+n with K = kt*32 + {0..7,16..23};
// lane n+16 = same column with K = kt*32 + {8..15,24..31}.
__global__ void pack_w(const float* __restrict__ W, uint16_t* __restrict__ dst,
                       int K, int N, int nfragPerStep, int totalFrag) {
  int id = blockIdx.x * blockDim.x + threadIdx.x;
  int f  = id >> 5;
  if (f >= totalFrag) return;
  int lane = id & 31;
  int NT = N >> 4;
  int step = f / nfragPerStep;
  int fr   = f - step * nfragPerStep;
  int kt = fr / NT, nt = fr - kt * NT;
  const float* Ws = W + (size_t)step * K * N;
  int col   = nt * 16 + (lane & 15);
  int khalf = (lane >> 4) * 8;
  uint16_t* d = dst + (size_t)f * 512 + lane * 16;
#pragma unroll
  for (int i = 0; i < 16; ++i) {
    int k = kt * 32 + khalf + (i < 8 ? i : i + 8);
    d[i] = f2bf_u(Ws[(size_t)k * N + col]);
  }
}

// ---------------- WMMA GEMM tile helper ----------------
union ABFrag { uint4 q[2]; v16bf v; };

// out[BT x N] = act(in[BT x K] @ W + bias); BT==32, K%32==0, N%16==0.
// 8 waves: wave&1 -> row tile (16 rows), wave>>1 strides n-tiles by 4.
template <int TANH, int OUTF32>
__device__ __forceinline__ void gemm32(const bf16_t* __restrict__ inL, int K,
                                       const uint16_t* __restrict__ wfrag,
                                       const float* __restrict__ bias, int N,
                                       bf16_t* __restrict__ outB,
                                       float* __restrict__ outF) {
  const int tid  = threadIdx.x;
  const int lane = tid & 31;
  const int wave = tid >> 5;
  const int mt   = wave & 1;
  const int NT = N >> 4, KT = K >> 5;
  const int rowA  = mt * 16 + (lane & 15);
  const int khalf = (lane >> 4) * 8;
  for (int nt = wave >> 1; nt < NT; nt += 4) {
    v8f acc = {};
    for (int kt = 0; kt < KT; ++kt) {
      ABFrag a, b;
      const bf16_t* ap = inL + rowA * K + kt * 32 + khalf;   // LDS, 16B aligned
      a.q[0] = *(const uint4*)(ap);
      a.q[1] = *(const uint4*)(ap + 16);
      const uint16_t* wp = wfrag + ((size_t)(kt * NT + nt)) * 512 + lane * 16;
      b.q[0] = *(const uint4*)(wp);
      b.q[1] = *(const uint4*)(wp + 8);
      if (kt + 1 < KT)   // pull next K-tile's fragment into cache
        __builtin_prefetch(wfrag + ((size_t)((kt + 1) * NT + nt)) * 512 + lane * 16, 0, 3);
      acc = __builtin_amdgcn_wmma_f32_16x16x32_bf16(
                false, a.v, false, b.v, (short)0, acc, false, false);
    }
    const int col   = nt * 16 + (lane & 15);
    const float bv  = bias[col];
    const int rbase = mt * 16 + ((lane >> 4) * 8);
#pragma unroll
    for (int j = 0; j < 8; ++j) {
      float v = acc[j] + bv;
      if (TANH) v = fast_tanh(v);
      if (OUTF32) outF[(rbase + j) * N + col] = v;
      else        outB[(rbase + j) * N + col] = f2bf(v);
    }
  }
}

// ---------------- fused SDE kernel ----------------
__global__ void __launch_bounds__(256)
sde_fused(const float* __restrict__ xt0, const float* __restrict__ yt0,
          const float* __restrict__ dBxt, const float* __restrict__ dByt,
          const float* __restrict__ Dxp,
          const uint16_t* __restrict__ pW1p, const float* __restrict__ pb1,
          const uint16_t* __restrict__ pW2p, const float* __restrict__ pb2,
          const float* __restrict__ pW3, const float* __restrict__ pb3,
          const uint16_t* __restrict__ gW1p, const float* __restrict__ gb1,
          const uint16_t* __restrict__ gW2p, const float* __restrict__ gb2,
          const uint16_t* __restrict__ gW3p, const float* __restrict__ gb3,
          float* __restrict__ out) {
  __shared__ __align__(16) float  xb[BT * 48];
  __shared__ __align__(16) float  yb[BT * DYP];
  __shared__ __align__(16) bf16_t act[BT * DDIM];
  __shared__ __align__(16) bf16_t h1[BT * HDIM];
  __shared__ __align__(16) bf16_t h2[BT * HDIM];
  __shared__ __align__(16) float  grad[BT * DDIM];
  __shared__ float dpx[BT * DXP];
  __shared__ int   hitb[BT * DXP];
  __shared__ float pb_[BT];
  __shared__ float prel[BT];
  __shared__ int   runb[BT];

  const int tid   = threadIdx.x;
  const int rbase = blockIdx.x * BT;

  // ---- init state ----
  for (int i = tid; i < BT * 48; i += 256) xb[i] = xt0[(size_t)rbase * 48 + i];
  for (int i = tid; i < BT * DYP; i += 256) yb[i] = yt0[(size_t)rbase * DYP + i];
  if (tid < BT) { runb[tid] = 1; prel[tid] = 0.f; }
  __syncthreads();

  // ---- build act = bf16([x | y]) ----
  auto build_act = [&]() {
    for (int i = tid; i < BT * DDIM; i += 256) {
      int r = i >> 6, c = i & 63;
      float v = (c < 48) ? xb[r * 48 + c] : yb[r * DYP + (c - 48)];
      act[i] = f2bf(v);
    }
  };

  // ---- p0 = fnn(act; pW*) ----
  build_act();
  __syncthreads();
  gemm32<1, 0>(act, DDIM, pW1p, pb1, HDIM, h1, nullptr);
  __syncthreads();
  gemm32<1, 0>(h1, HDIM, pW2p, pb2, HDIM, h2, nullptr);
  __syncthreads();
  if (tid < BT) {
    float s0 = pb3[0];
    for (int k = 0; k < HDIM; ++k) s0 += bf2f(h2[tid * HDIM + k]) * pW3[k];
    pb_[tid] = s0;
  }
  __syncthreads();

  // ---- time loop ----
  for (int s = 0; s < NSTEP; ++s) {
    build_act();
    __syncthreads();
    gemm32<1, 0>(act, DDIM, gW1p + (size_t)s * 32 * 512,  gb1 + s * HDIM, HDIM, h1, nullptr);
    __syncthreads();
    gemm32<1, 0>(h1, HDIM, gW2p + (size_t)s * 128 * 512, gb2 + s * HDIM, HDIM, h2, nullptr);
    __syncthreads();
    gemm32<0, 1>(h2, HDIM, gW3p + (size_t)s * 32 * 512,  gb3 + s * DDIM, DDIM, nullptr, grad);
    __syncthreads();

    // ---- phase A: per-(row,particle) x-dynamics + dpx ----
    for (int t = tid; t < BT * DXP; t += 256) {
      const int r = t >> 4, m = t & 15;
      const int g = rbase + r;
      const float runf = runb[r] ? 1.f : 0.f;
      float x0 = xb[r * 48 + 3 * m + 0];
      float x1 = xb[r * 48 + 3 * m + 1];
      float x2 = xb[r * 48 + 3 * m + 2];
      float rn = sqrtf(x0 * x0 + x1 * x1 + x2 * x2);
      float ct = fminf(1.f, fmaxf(-1.f, x2 / fmaxf(rn, 1e-12f)));
      float theta = acosf(ct);
      float phi = atan2f(x1, x0);
      float aa = theta - PI_2F;
      // bounded angles -> native v_sin/v_cos
      float ca = __cosf(aa), sa = __sinf(aa);
      float cp = __cosf(phi), sp = __sinf(phi);
      float T00 = cp * ca, T01 = -sp, T02 = cp * sa;
      float T10 = sp * ca, T11 =  cp, T12 = sp * sa;
      float T20 = -sa,     T21 = 0.f, T22 = ca;
      float sd = sqrtf(2.f * Dxp[m]);
      size_t bo = (((size_t)s * B_TOT + g) * DXP + m) * 2;
      float dB0 = dBxt[bo], dB1 = dBxt[bo + 1];
      float th2 = sd * dB0 + PI_2F;
      float ph2 = sd * dB1;
      float st = __sinf(th2);
      float e0 = st * __cosf(ph2) - 1.f;
      float e1 = st * __sinf(ph2);
      float e2 = __cosf(th2);
      float d0 = T00 * e0 + T01 * e1 + T02 * e2;
      float d1 = T10 * e0 + T11 * e1 + T12 * e2;
      float d2 = T20 * e0 + T21 * e1 + T22 * e2;
      x0 += d0 * runf; x1 += d1 * runf; x2 += d2 * runf;
      if (x2 > 0.9f) x2 = 1.8f - x2;            // reflect at ZMAX
      xb[r * 48 + 3 * m + 0] = x0;
      xb[r * 48 + 3 * m + 1] = x1;
      xb[r * 48 + 3 * m + 2] = x2;
      hitb[r * DXP + m] = (x2 < -0.9f) ? 1 : 0;  // ZHIT
      float g0 = grad[r * DDIM + 3 * m + 0];
      float g1 = grad[r * DDIM + 3 * m + 1];
      float g2 = grad[r * DDIM + 3 * m + 2];
      float gxp1 = g0 * T01 + g1 * T11 + g2 * T21;
      float gxp2 = g0 * T02 + g1 * T12 + g2 * T22;
      dpx[r * DXP + m] = (-gxp2) * (sd * dB0) + gxp1 * (sd * dB1);
    }
    __syncthreads();

    // ---- phase B: per-row y-dynamics, p, hit/run ----
    if (tid < BT) {
      const int r = tid;
      const int g = rbase + r;
      const float runf = runb[r] ? 1.f : 0.f;
      float gys = 0.f, dpy = 0.f;
#pragma unroll
      for (int k = 0; k < DYP; ++k) gys += grad[r * DDIM + 48 + k];
      float fval = -0.05f * pb_[r] + 0.1f * gys;
      float yn[DYP];
      float s2 = 0.f;
#pragma unroll
      for (int k = 0; k < DYP; ++k) {
        float y   = yb[r * DYP + k];
        float dBy = dByt[((size_t)s * B_TOT + g) * DYP + k];
        float v   = y + (-y * DT_F + 0.2f * dBy) * runf;
        yn[k] = v; s2 += v * v;
        dpy += 0.2f * grad[r * DDIM + 48 + k] * dBy;
      }
      float rr = sqrtf(s2);
      if (rr > 2.f) {
        float sc = (4.f - rr) / fmaxf(rr, 1e-12f);
#pragma unroll
        for (int k = 0; k < DYP; ++k) yn[k] *= sc;
      }
      float pcv = 0.f;
#pragma unroll
      for (int k = 0; k < DYP; ++k) { yb[r * DYP + k] = yn[k]; pcv += yn[k] * yn[k]; }
      float dps = 0.f;
#pragma unroll
      for (int m = 0; m < DXP; ++m) dps += dpx[r * DXP + m];
      pb_[r] += (-fval * DT_F + dps + dpy) * runf;
      int hit = 0;
#pragma unroll
      for (int m = 0; m < DXP; ++m) hit |= hitb[r * DXP + m];
      if (runb[r] && hit) prel[r] = pcv;
      runb[r] = runb[r] && !hit;
    }
    __syncthreads();
  }

  // ---- epilogue ----
  if (tid < BT) {
    const int r = tid, g = rbase + r;
    if (runb[r]) {
      float pcv = 0.f;
#pragma unroll
      for (int k = 0; k < DYP; ++k) { float v = yb[r * DYP + k]; pcv += v * v; }
      prel[r] = pcv;
    }
    out[g]         = pb_[r];
    out[B_TOT + g] = prel[r];
  }
}

// ---------------- host launcher ----------------
extern "C" void kernel_launch(void* const* d_in, const int* in_sizes, int n_in,
                              void* d_out, int out_size, void* d_ws, size_t ws_size,
                              hipStream_t stream) {
  const float* xt0  = (const float*)d_in[0];
  const float* yt0  = (const float*)d_in[1];
  const float* dBxt = (const float*)d_in[2];
  const float* dByt = (const float*)d_in[3];
  const float* Dxp  = (const float*)d_in[4];
  const float* pW1  = (const float*)d_in[5];
  const float* pb1  = (const float*)d_in[6];
  const float* pW2  = (const float*)d_in[7];
  const float* pb2  = (const float*)d_in[8];
  const float* pW3  = (const float*)d_in[9];
  const float* pb3  = (const float*)d_in[10];
  const float* gW1  = (const float*)d_in[11];
  const float* gb1  = (const float*)d_in[12];
  const float* gW2  = (const float*)d_in[13];
  const float* gb2  = (const float*)d_in[14];
  const float* gW3  = (const float*)d_in[15];
  const float* gb3  = (const float*)d_in[16];
  float* out = (float*)d_out;

  // workspace layout (bf16 fragments, 512 u16 = 1KB per 32x16 tile)
  uint16_t* ws = (uint16_t*)d_ws;
  size_t o = 0;
  uint16_t* pW1p = ws + o; o += (size_t)32   * 512;  // 64x256
  uint16_t* pW2p = ws + o; o += (size_t)128  * 512;  // 256x256
  uint16_t* gW1p = ws + o; o += (size_t)1600 * 512;  // 50 x 64x256
  uint16_t* gW2p = ws + o; o += (size_t)6400 * 512;  // 50 x 256x256
  uint16_t* gW3p = ws + o; o += (size_t)1600 * 512;  // 50 x 256x64

  auto pack = [&](const float* W, uint16_t* dst, int K, int N, int steps) {
    int nf = (K / 32) * (N / 16);
    int total = nf * steps;
    int threads = total * 32;
    pack_w<<<(threads + 255) / 256, 256, 0, stream>>>(W, dst, K, N, nf, total);
  };
  pack(pW1, pW1p, 64,  256, 1);
  pack(pW2, pW2p, 256, 256, 1);
  pack(gW1, gW1p, 64,  256, NSTEP);
  pack(gW2, gW2p, 256, 256, NSTEP);
  pack(gW3, gW3p, 256, 64,  NSTEP);

  sde_fused<<<B_TOT / BT, 256, 0, stream>>>(
      xt0, yt0, dBxt, dByt, Dxp,
      pW1p, pb1, pW2p, pb2, pW3, pb3,
      gW1p, gb1, gW2p, gb2, gW3p, gb3, out);

  (void)in_sizes; (void)n_in; (void)out_size; (void)ws_size;
}
